// Linear_25967372272130
// MI455X (gfx1250) — compile-verified, matrix operand order checked by reading, not run on hardware
//
#include <hip/hip_runtime.h>
#include <hip/hip_bf16.h>
#include <hip/hip_fp16.h>
#include <stdint.h>

// ---------------------------------------------------------------------------
// y[m][n] = scale[n] * sum_k x[m][k] * (float)q[n][k]
// int8 q in [-31,31] (int6) -> exact bf16 via bit trick; scale in f32 epilogue.
// A tile staged via global_load_async_to_lds_b128 (ASYNCcnt);
// B tile dequantized with v_perm_b32 + v_pk_add_bf16 magic-number trick.
// M = 16384, N = 4096, K = 4096
// ---------------------------------------------------------------------------

typedef __attribute__((ext_vector_type(16))) __bf16 v16bf;
typedef __attribute__((ext_vector_type(8)))  __bf16 v8bf;
typedef __attribute__((ext_vector_type(2)))  __bf16 v2bf;
typedef __attribute__((ext_vector_type(8)))  float  v8f;

#define K_DIM 4096
#define N_DIM 4096
#define BM 256
#define BN 128
#define BK 32
#define LDSS 40   // padded row stride in bf16 (80B = 20 banks; 20*l mod 64 distinct)
#define NT (K_DIM / BK)

__device__ __forceinline__ void async_copy_row64B(uint32_t lds_addr, const void* gptr) {
    // 64 contiguous bytes global -> LDS (4 x b128).
    // INST_OFFSET is added to BOTH the memory and the LDS address (ISA 10.x).
    asm volatile(
        "global_load_async_to_lds_b128 %0, %1, off\n\t"
        "global_load_async_to_lds_b128 %0, %1, off offset:16\n\t"
        "global_load_async_to_lds_b128 %0, %1, off offset:32\n\t"
        "global_load_async_to_lds_b128 %0, %1, off offset:48"
        :: "v"(lds_addr), "v"(gptr) : "memory");
}

__device__ __forceinline__ void wait_async_done() {
    asm volatile("s_wait_asynccnt 0x0" ::: "memory");
}

// Exact int6 -> bf16 dequant, 16 elements (4 dwords in, 8 dwords of packed bf16 out).
// v6 = (e+32) & 63 in [1,63] fits bf16's 7-bit mantissa: bits(0x4300|v6) = 128+v6.
// (128+v6) - 160 = e, exact. -160 in bf16 = 0xC320.
__device__ __forceinline__ void dequant16_bf16(const int4& b, uint32_t* o /*[8]*/) {
    const uint32_t* w = (const uint32_t*)&b;
    const v2bf magic = {(__bf16)(-160.0f), (__bf16)(-160.0f)};
#pragma unroll
    for (int c = 0; c < 4; ++c) {
        const uint32_t v = ((w[c] & 0x3F3F3F3Fu) + 0x20202020u) & 0x3F3F3F3Fu;
        const uint32_t lo = __builtin_amdgcn_perm(0x43434343u, v, 0x04010400u); // {43,b1,43,b0}
        const uint32_t hi = __builtin_amdgcn_perm(0x43434343u, v, 0x04030402u); // {43,b3,43,b2}
        v2bf r0 = __builtin_bit_cast(v2bf, lo) + magic;  // v_pk_add_bf16
        v2bf r1 = __builtin_bit_cast(v2bf, hi) + magic;
        o[2 * c + 0] = __builtin_bit_cast(uint32_t, r0);
        o[2 * c + 1] = __builtin_bit_cast(uint32_t, r1);
    }
}

__global__ __launch_bounds__(256) void Linear_25967372272130_kernel(
    const __hip_bfloat16* __restrict__ x,
    const int8_t* __restrict__ q,
    const __half* __restrict__ scale,
    __hip_bfloat16* __restrict__ out)
{
    // ping-pong tiles: A = 256x32 bf16, B = 128x32 bf16 (dequantized)
    __shared__ __align__(16) __bf16 As[2][BM * LDSS];
    __shared__ __align__(16) __bf16 Bs[2][BN * LDSS];

    const int tid   = threadIdx.x;
    const int lane  = tid & 31;
    const int wave  = tid >> 5;
    const int waveM = (wave & 3) * 64;   // 4 waves tile M (64 rows each)
    const int waveN = (wave >> 2) * 64;  // 2 waves tile N (64 cols each)

    const int m0 = blockIdx.y * BM;
    const int n0 = blockIdx.x * BN;

    // WMMA 16-bit operand lane mapping: row = lane&15, K phase = (lane>>4)*8
    const int lrow = lane & 15;
    const int lk   = (lane >> 4) * 8;

    v8f acc[4][4];
    const v8f vzero = {0.f, 0.f, 0.f, 0.f, 0.f, 0.f, 0.f, 0.f};
#pragma unroll
    for (int i = 0; i < 4; ++i)
#pragma unroll
        for (int j = 0; j < 4; ++j)
            acc[i][j] = vzero;

    // A staging: 1 row (32 bf16 = 64B) per thread, async global->LDS
    const int rowA = tid;
    // B staging: 2 threads per row, 16 int8 (= 1x int4) each
    const int rowB  = tid >> 1;
    const int koffB = (tid & 1) * 16;

    const __hip_bfloat16* xrow = x + (size_t)(m0 + rowA) * K_DIM;
    const int8_t*         qrow = q + (size_t)(n0 + rowB) * K_DIM + koffB;

    const uint32_t ldsA0 = (uint32_t)(uintptr_t)(&As[0][rowA * LDSS]);
    const uint32_t ldsA1 = (uint32_t)(uintptr_t)(&As[1][rowA * LDSS]);

    // ---- preload tile 0 into buffer 0
    {
        async_copy_row64B(ldsA0, xrow);

        int4 b0 = *(const int4*)(qrow);
        __align__(16) uint32_t tmp[8];
        dequant16_bf16(b0, tmp);
        int4* bsp = (int4*)(&Bs[0][rowB * LDSS + koffB]);
        bsp[0] = ((int4*)tmp)[0];
        bsp[1] = ((int4*)tmp)[1];

        wait_async_done();
    }
    __syncthreads();

    for (int t = 0; t < NT; ++t) {
        const int p  = t & 1;
        const int kt = (t + 1) * BK;
        const bool more = (t + 1 < NT);

        // ---- issue next-tile loads early (latency hidden by compute below)
        int4 b0;
        if (more) {
            async_copy_row64B(p ? ldsA0 : ldsA1, xrow + kt);   // A: direct to LDS
            b0 = *(const int4*)(qrow + kt);                    // B: via VGPRs (dequant)
        }

        // ---- compute tile t from buffer p: 16 WMMAs
        {
            v16bf afrag[4], bfrag[4];
#pragma unroll
            for (int i = 0; i < 4; ++i) {
                const __bf16* pa = &As[p][(waveM + i * 16 + lrow) * LDSS + lk];
                v8bf lo = *(const v8bf*)pa;
                v8bf hi = *(const v8bf*)(pa + 16);
                afrag[i] = __builtin_shufflevector(lo, hi,
                    0, 1, 2, 3, 4, 5, 6, 7, 8, 9, 10, 11, 12, 13, 14, 15);
            }
#pragma unroll
            for (int j = 0; j < 4; ++j) {
                const __bf16* pb = &Bs[p][(waveN + j * 16 + lrow) * LDSS + lk];
                v8bf lo = *(const v8bf*)pb;
                v8bf hi = *(const v8bf*)(pb + 16);
                bfrag[j] = __builtin_shufflevector(lo, hi,
                    0, 1, 2, 3, 4, 5, 6, 7, 8, 9, 10, 11, 12, 13, 14, 15);
            }
#pragma unroll
            for (int i = 0; i < 4; ++i)
#pragma unroll
                for (int j = 0; j < 4; ++j)
                    acc[i][j] = __builtin_amdgcn_wmma_f32_16x16x32_bf16(
                        /*neg_a=*/false, afrag[i],
                        /*neg_b=*/false, bfrag[j],
                        /*c_mod=*/(short)0, acc[i][j],
                        /*reuse_a=*/false, /*reuse_b=*/false);
        }

        // ---- dequant + stage B tile t+1 into buffer p^1
        if (more) {
            __align__(16) uint32_t tmp[8];
            dequant16_bf16(b0, tmp);
            int4* bsp = (int4*)(&Bs[p ^ 1][rowB * LDSS + koffB]);
            bsp[0] = ((int4*)tmp)[0];
            bsp[1] = ((int4*)tmp)[1];

            wait_async_done();   // A tile t+1 fully in LDS before publishing
        }
        __syncthreads();
    }

    // ---- epilogue: apply scale[n] in f32, convert to bf16, store
    // C/D layout: VGPR e holds M = e + 8*(lane>>4), lane&15 = N
    const int mphase = (lane >> 4) * 8;
#pragma unroll
    for (int j = 0; j < 4; ++j) {
        const int n = n0 + waveN + j * 16 + lrow;
        const float sc = __half2float(scale[n]);
#pragma unroll
        for (int i = 0; i < 4; ++i) {
#pragma unroll
            for (int e = 0; e < 8; ++e) {
                const int m = m0 + waveM + i * 16 + mphase + e;
                out[(size_t)m * N_DIM + n] = __float2bfloat16(acc[i][j][e] * sc);
            }
        }
    }
}

extern "C" void kernel_launch(void* const* d_in, const int* in_sizes, int n_in,
                              void* d_out, int out_size, void* d_ws, size_t ws_size,
                              hipStream_t stream) {
    (void)n_in; (void)out_size; (void)d_ws; (void)ws_size;

    const __hip_bfloat16* x     = (const __hip_bfloat16*)d_in[0];
    const int8_t*         qw    = (const int8_t*)d_in[1];
    const __half*         scale = (const __half*)d_in[2];
    __hip_bfloat16*       out   = (__hip_bfloat16*)d_out;

    const int M = in_sizes[0] / K_DIM;   // 8*2048 = 16384

    dim3 grid(N_DIM / BN, M / BM);       // (32, 64)
    dim3 block(256);
    Linear_25967372272130_kernel<<<grid, block, 0, stream>>>(x, qw, scale, out);
}
